// NERLSTM_CRF_19585050870487
// MI455X (gfx1250) — matrix operand order repaired and checked by reading, not compiled, not run on hardware
//
#include <hip/hip_runtime.h>
#include <hip/hip_bf16.h>

typedef _Float16 half_t;
typedef __attribute__((ext_vector_type(16))) _Float16 v16h;
typedef __attribute__((ext_vector_type(8)))  _Float16 h8;
typedef __attribute__((ext_vector_type(8)))  float    v8f;
typedef __attribute__((ext_vector_type(4)))  unsigned int u32x4;
typedef __attribute__((ext_vector_type(8)))  int      i32x8;
typedef __attribute__((ext_vector_type(4)))  int      i32x4;

#define SEQ   256
#define BATCH 256
#define EDIM  300
#define HDIM  300
#define EP    320   // E padded to multiple of 32 (K tiles)
#define HP    320   // H padded to multiple of 32
#define G4    1200  // 4*H gate columns
#define NTILE 75    // G4 / 16
#define KT_E  10    // EP / 32
#define KT_H  10    // HP / 32
#define NTAGS 9
#define HS    304   // h storage row stride (16B aligned in f16)
#define EMS   16    // emission row stride (floats)

#if defined(__HIP_DEVICE_COMPILE__) && __has_builtin(__builtin_amdgcn_tensor_load_to_lds) && \
    __has_builtin(__builtin_amdgcn_s_wait_tensorcnt)
#define USE_TDM 1
#else
#define USE_TDM 0
#endif

__device__ __forceinline__ float sigf(float x) { return 1.f / (1.f + __expf(-x)); }

// A-fragment (16x32 f16): lane = row (lane&15); half = lane>>4.
// In-lane half order: K = half*16 + {0..15} (contiguous).
__device__ __forceinline__ v16h load_a_frag(const half_t* rowbase, int koff) {
  h8 lo = *(const h8*)(rowbase + koff);
  h8 hi = *(const h8*)(rowbase + koff + 8);
  v16h a;
#pragma unroll
  for (int i = 0; i < 8; ++i) { a[i] = lo[i]; a[i + 8] = hi[i]; }
  return a;
}

#if USE_TDM
// 1-D TDM copy: 10240 contiguous bytes (16 rows x EP f16) global -> LDS.
// D# group0: count=1 | lds_addr[63:32] | global_addr[120:64] | type=2.
// D# group1: data_size=8B, tensor_dim0=tile_dim0=1280, dim1=1, stride0=1280.
__device__ __forceinline__ void tdm_load_xtile(const half_t* gsrc, void* ldsdst) {
  unsigned long long ga = (unsigned long long)(uintptr_t)gsrc;
  unsigned lds = (unsigned)(uintptr_t)ldsdst;   // low 32 bits of generic = LDS offset
  u32x4 g0 = { 1u, lds, (unsigned)ga,
               (unsigned)((ga >> 32) & 0x01FFFFFFu) | 0x80000000u };
  i32x8 g1 = { 0x30000,          // data_size=3 (8B), mask=0, no pad/iterate/barrier
               0x05000000,       // tensor_dim0 lo16 (=1280) in bits [63:48]
               0x00010000,       // tensor_dim0 hi16=0, tensor_dim1 lo16=1
               0x05000000,       // tensor_dim1 hi16=0, tile_dim0=1280
               1,                // tile_dim1=1, tile_dim2=0
               1280,             // tensor_dim0_stride lo32
               0, 0 };
  i32x4 gz = { 0, 0, 0, 0 };
#if __clang_major__ >= 23
  i32x8 gz8 = { 0, 0, 0, 0, 0, 0, 0, 0 };
  __builtin_amdgcn_tensor_load_to_lds(g0, g1, gz, gz, gz8, 0);
#else
  __builtin_amdgcn_tensor_load_to_lds(g0, g1, gz, gz, 0);
#endif
}
#endif

// ---------------------------------------------------------------------------
// 1) Embedding gather: xs[t][b][e] = emb[x[b][t]][e], f16, E padded to EP.
// ---------------------------------------------------------------------------
__global__ __launch_bounds__(256) void embed_kernel(const int* __restrict__ x,
                                                    const float* __restrict__ emb,
                                                    half_t* __restrict__ xs) {
  int idx = blockIdx.x * 256 + threadIdx.x;          // < SEQ*BATCH*EP
  int e = idx % EP;
  int r = idx / EP;
  int b = r % BATCH;
  int s = r / BATCH;
  int tok = x[b * SEQ + s];
  float v = (e < EDIM) ? emb[(size_t)tok * EDIM + e] : 0.f;
  xs[(size_t)idx] = (half_t)v;
}

// ---------------------------------------------------------------------------
// 2) Weight convert f32 [rows][cols] -> f16 [rows][cols_pad] (zero padded).
// ---------------------------------------------------------------------------
__global__ __launch_bounds__(256) void wconv_kernel(const float* __restrict__ w,
                                                    half_t* __restrict__ out,
                                                    int rows, int cols, int cols_pad) {
  int idx = blockIdx.x * 256 + threadIdx.x;
  if (idx >= rows * cols_pad) return;
  int c = idx % cols_pad;
  int r = idx / cols_pad;
  out[idx] = (half_t)((c < cols) ? w[(size_t)r * cols + c] : 0.f);
}

__global__ __launch_bounds__(256) void bias_kernel(const float* __restrict__ a,
                                                   const float* __restrict__ b,
                                                   float* __restrict__ out, int n) {
  int i = blockIdx.x * 256 + threadIdx.x;
  if (i < n) out[i] = a[i] + b[i];
}

// ---------------------------------------------------------------------------
// 3) Persistent WMMA BiLSTM. grid = (BATCH/16, 2 directions), block = 256.
//    h,c resident in LDS; x_t tiles double-buffered via TDM (tensor_load_to_lds
//    + s_wait_tensorcnt) so the t+1 copy overlaps the 20-WMMA/N-tile compute.
// ---------------------------------------------------------------------------
__global__ __launch_bounds__(256) void bilstm_kernel(
    const half_t* __restrict__ xs,
    const half_t* __restrict__ wihF, const half_t* __restrict__ whhF,
    const float* __restrict__ biasF,
    const half_t* __restrict__ wihB, const half_t* __restrict__ whhB,
    const float* __restrict__ biasB,
    half_t* __restrict__ hF, half_t* __restrict__ hB) {

  __shared__ __align__(16) half_t xA[2][16][EP];  // 20.0 KB  double-buffered x tile
  __shared__ __align__(16) half_t hA[16][HP];     // 10.0 KB  h_{t-1} tile (A operand)
  __shared__ __align__(16) float  gbuf[16][G4];   // 76.8 KB  gate pre-activations
  __shared__ __align__(16) float  cbuf[16][HDIM]; // 19.2 KB  cell state
  __shared__ __align__(16) float  bsh[G4];        //  4.8 KB  fused bias

  const int tid     = threadIdx.x;
  const int lane    = tid & 31;
  const int wave    = tid >> 5;
  const int row     = lane & 15;
  const int half_id = lane >> 4;
  const int dir     = blockIdx.y;
  const int bglob   = blockIdx.x * 16;

  const half_t* Wih  = dir ? wihB : wihF;
  const half_t* Whh  = dir ? whhB : whhF;
  const float*  bs   = dir ? biasB : biasF;
  half_t*       hout = dir ? hB : hF;

  for (int i = tid; i < G4; i += 256) bsh[i] = bs[i];
  for (int i = tid; i < 16 * HP; i += 256) (&hA[0][0])[i] = (half_t)0.f;
  for (int i = tid; i < 16 * HDIM; i += 256) (&cbuf[0][0])[i] = 0.f;

#if USE_TDM
  // Preload x tile for step 0 into buffer 0 (one TDM issue from wave 0).
  if (wave == 0) {
    const int t0 = dir ? (SEQ - 1) : 0;
    tdm_load_xtile(xs + ((size_t)t0 * BATCH + bglob) * EP, &xA[0][0][0]);
  }
#endif
  __syncthreads();

  for (int step = 0; step < SEQ; ++step) {
    const int t   = dir ? (SEQ - 1 - step) : step;
    const int cur = step & 1;

#if USE_TDM
    if (wave == 0) {
      if (step + 1 < SEQ) {
        const int tn = dir ? (SEQ - 2 - step) : (step + 1);
        tdm_load_xtile(xs + ((size_t)tn * BATCH + bglob) * EP, &xA[cur ^ 1][0][0]);
        __builtin_amdgcn_s_wait_tensorcnt(1);  // tile for step t has landed
      } else {
        __builtin_amdgcn_s_wait_tensorcnt(0);
      }
    }
    __syncthreads();
#else
    {
      const uint4* s4 = (const uint4*)(xs + ((size_t)t * BATCH + bglob) * EP);
      uint4* d4 = (uint4*)(&xA[cur][0][0]);
      for (int i = tid; i < (16 * EP) / 8; i += 256) d4[i] = s4[i];
      __syncthreads();
    }
#endif

    // --- gate GEMM: gbuf[16][1200] = x_t*Wih^T + h*Whh^T ---
    for (int nt = wave; nt < NTILE; nt += 8) {
      if (nt + 8 < NTILE) {
        __builtin_prefetch(Wih + (size_t)(nt + 8) * 16 * EP, 0, 1);
        __builtin_prefetch(Whh + (size_t)(nt + 8) * 16 * HP, 0, 1);
      }
      const int nrow = nt * 16 + row;          // weight row = gate column
      v8f acc = {};
#pragma unroll
      for (int k = 0; k < KT_E; ++k) {
        v16h a = load_a_frag(&xA[cur][row][0], k * 32 + half_id * 16);
        v16h b = *(const v16h*)(Wih + (size_t)nrow * EP + k * 32 + half_id * 16);
        acc = __builtin_amdgcn_wmma_f32_16x16x32_f16(false, a, false, b,
                                                     (short)0, acc, false, false);
      }
#pragma unroll
      for (int k = 0; k < KT_H; ++k) {
        v16h a = load_a_frag(&hA[row][0], k * 32 + half_id * 16);
        v16h b = *(const v16h*)(Whh + (size_t)nrow * HP + k * 32 + half_id * 16);
        acc = __builtin_amdgcn_wmma_f32_16x16x32_f16(false, a, false, b,
                                                     (short)0, acc, false, false);
      }
      // C/D layout: VGPR v -> row v + 8*half, col lane&15
#pragma unroll
      for (int v = 0; v < 8; ++v)
        gbuf[v + 8 * half_id][nt * 16 + row] = acc[v];
    }
    __syncthreads();

    // --- elementwise LSTM cell (PyTorch gate order i,f,g,o) ---
    for (int idx = tid; idx < 16 * HDIM; idx += 256) {
      int bb = idx / HDIM, jj = idx % HDIM;
      float gi = gbuf[bb][jj]            + bsh[jj];
      float gf = gbuf[bb][HDIM + jj]     + bsh[HDIM + jj];
      float gg = gbuf[bb][2 * HDIM + jj] + bsh[2 * HDIM + jj];
      float go = gbuf[bb][3 * HDIM + jj] + bsh[3 * HDIM + jj];
      float c = sigf(gf) * cbuf[bb][jj] + sigf(gi) * tanhf(gg);
      cbuf[bb][jj] = c;
      float h = sigf(go) * tanhf(c);
      hA[bb][jj] = (half_t)h;
      hout[((size_t)t * BATCH + bglob + bb) * HS + jj] = (half_t)h;
    }
    __syncthreads();
  }
}

// ---------------------------------------------------------------------------
// 4) Emissions: em[s][b][t] = [hF|hB] @ w_proj^T + b_proj  (T=9: scalar FMA)
// ---------------------------------------------------------------------------
__global__ __launch_bounds__(256) void emission_kernel(
    const half_t* __restrict__ hF, const half_t* __restrict__ hB,
    const float* __restrict__ wproj, const float* __restrict__ bproj,
    float* __restrict__ em) {
  __shared__ float wp[NTAGS][2 * HDIM];   // 21.6 KB
  int tid = threadIdx.x;
  for (int i = tid; i < NTAGS * 2 * HDIM; i += 256) (&wp[0][0])[i] = wproj[i];
  __syncthreads();

  int gid = blockIdx.x * 256 + tid;       // (s,b) flat == s*BATCH + b
  const half_t* pf = hF + (size_t)gid * HS;
  const half_t* pb = hB + (size_t)gid * HS;
  float acc[NTAGS];
#pragma unroll
  for (int tg = 0; tg < NTAGS; ++tg) acc[tg] = bproj[tg];
  for (int j = 0; j < HDIM; ++j) {
    float hf = (float)pf[j];
#pragma unroll
    for (int tg = 0; tg < NTAGS; ++tg) acc[tg] += hf * wp[tg][j];
  }
  for (int j = 0; j < HDIM; ++j) {
    float hb = (float)pb[j];
#pragma unroll
    for (int tg = 0; tg < NTAGS; ++tg) acc[tg] += hb * wp[tg][HDIM + j];
  }
#pragma unroll
  for (int tg = 0; tg < NTAGS; ++tg) em[(size_t)gid * EMS + tg] = acc[tg];
}

// ---------------------------------------------------------------------------
// 5) CRF Viterbi: 1 block, thread = batch lane. max-plus DP + backtrack.
// ---------------------------------------------------------------------------
__global__ __launch_bounds__(256) void viterbi_kernel(
    const float* __restrict__ em, const float* __restrict__ start_t,
    const float* __restrict__ end_t, const float* __restrict__ trans,
    unsigned char* __restrict__ bp, int* __restrict__ out) {
  __shared__ float tr[NTAGS][NTAGS];
  __shared__ float st[NTAGS], et[NTAGS];
  int tid = threadIdx.x;
  if (tid < NTAGS * NTAGS) (&tr[0][0])[tid] = trans[tid];
  if (tid < NTAGS) { st[tid] = start_t[tid]; et[tid] = end_t[tid]; }
  __syncthreads();

  int b = tid;
  float score[NTAGS];
#pragma unroll
  for (int tg = 0; tg < NTAGS; ++tg)
    score[tg] = st[tg] + em[(size_t)b * EMS + tg];

  for (int s = 1; s < SEQ; ++s) {
    const float* e = em + ((size_t)s * BATCH + b) * EMS;
    unsigned char* bprow = bp + ((size_t)s * BATCH + b) * 16;
    float ns[NTAGS];
#pragma unroll
    for (int tg = 0; tg < NTAGS; ++tg) {
      float best = score[0] + tr[0][tg];
      int bi = 0;
#pragma unroll
      for (int p = 1; p < NTAGS; ++p) {
        float v = score[p] + tr[p][tg];
        if (v > best) { best = v; bi = p; }   // strict > : first-max like argmax
      }
      ns[tg] = best + e[tg];
      bprow[tg] = (unsigned char)bi;
    }
#pragma unroll
    for (int tg = 0; tg < NTAGS; ++tg) score[tg] = ns[tg];
  }

  int last = 0;
  float best = score[0] + et[0];
#pragma unroll
  for (int tg = 1; tg < NTAGS; ++tg) {
    float v = score[tg] + et[tg];
    if (v > best) { best = v; last = tg; }
  }
  out[b * SEQ + (SEQ - 1)] = last;
  for (int s = SEQ - 2; s >= 0; --s) {
    last = bp[((size_t)(s + 1) * BATCH + b) * 16 + last];
    out[b * SEQ + s] = last;
  }
}

// ---------------------------------------------------------------------------
extern "C" void kernel_launch(void* const* d_in, const int* in_sizes, int n_in,
                              void* d_out, int out_size, void* d_ws, size_t ws_size,
                              hipStream_t stream) {
  (void)in_sizes; (void)n_in; (void)out_size; (void)ws_size;
  const int*   x      = (const int*)  d_in[0];
  const float* emb    = (const float*)d_in[1];
  const float* w_ih_f = (const float*)d_in[2];
  const float* w_hh_f = (const float*)d_in[3];
  const float* b_ih_f = (const float*)d_in[4];
  const float* b_hh_f = (const float*)d_in[5];
  const float* w_ih_b = (const float*)d_in[6];
  const float* w_hh_b = (const float*)d_in[7];
  const float* b_ih_b = (const float*)d_in[8];
  const float* b_hh_b = (const float*)d_in[9];
  const float* w_proj = (const float*)d_in[10];
  const float* b_proj = (const float*)d_in[11];
  const float* start_trans = (const float*)d_in[12];
  const float* end_trans   = (const float*)d_in[13];
  const float* trans  = (const float*)d_in[14];

  char* ws = (char*)d_ws;
  size_t off = 0;
  auto take = [&](size_t bytes) -> char* {
    char* p = ws + off;
    off += (bytes + 255) & ~(size_t)255;
    return p;
  };

  half_t* xs   = (half_t*)take((size_t)SEQ * BATCH * EP * 2);   // 41.9 MB
  half_t* wihF = (half_t*)take((size_t)G4 * EP * 2);
  half_t* whhF = (half_t*)take((size_t)G4 * HP * 2);
  half_t* wihB = (half_t*)take((size_t)G4 * EP * 2);
  half_t* whhB = (half_t*)take((size_t)G4 * HP * 2);
  float*  bF   = (float*) take((size_t)G4 * 4);
  float*  bB   = (float*) take((size_t)G4 * 4);
  half_t* hFp  = (half_t*)take((size_t)SEQ * BATCH * HS * 2);   // 39.8 MB
  half_t* hBp  = (half_t*)take((size_t)SEQ * BATCH * HS * 2);   // 39.8 MB
  float*  em   = (float*) take((size_t)SEQ * BATCH * EMS * 4);  //  4.2 MB
  unsigned char* bp = (unsigned char*)take((size_t)SEQ * BATCH * 16);

  embed_kernel<<<(SEQ * BATCH * EP) / 256, 256, 0, stream>>>(x, emb, xs);

  int wtot = G4 * EP;
  wconv_kernel<<<(wtot + 255) / 256, 256, 0, stream>>>(w_ih_f, wihF, G4, EDIM, EP);
  wconv_kernel<<<(wtot + 255) / 256, 256, 0, stream>>>(w_hh_f, whhF, G4, HDIM, HP);
  wconv_kernel<<<(wtot + 255) / 256, 256, 0, stream>>>(w_ih_b, wihB, G4, EDIM, EP);
  wconv_kernel<<<(wtot + 255) / 256, 256, 0, stream>>>(w_hh_b, whhB, G4, HDIM, HP);
  bias_kernel<<<(G4 + 255) / 256, 256, 0, stream>>>(b_ih_f, b_hh_f, bF, G4);
  bias_kernel<<<(G4 + 255) / 256, 256, 0, stream>>>(b_ih_b, b_hh_b, bB, G4);

  bilstm_kernel<<<dim3(BATCH / 16, 2), 256, 0, stream>>>(
      xs, wihF, whhF, bF, wihB, whhB, bB, hFp, hBp);

  emission_kernel<<<(SEQ * BATCH) / 256, 256, 0, stream>>>(hFp, hBp, w_proj, b_proj, em);

  viterbi_kernel<<<1, BATCH, 0, stream>>>(em, start_trans, end_trans, trans, bp,
                                          (int*)d_out);
}
